// RWKVRegionCell_12584254177391
// MI455X (gfx1250) — matrix operand order repaired and checked by reading, not compiled, not run on hardware
//
#include <hip/hip_runtime.h>
#include <hip/hip_bf16.h>

// ---------- problem constants ----------
#define D_DIM  1024
#define B_DIM  64
#define T_DIM  256
#define TBROWS (T_DIM * B_DIM)   // 16384
#define BD     (B_DIM * D_DIM)   // 65536
#define MTIME  16
#define KWTA_K 128
#define EPS_DIV 1e-6f
#define MAX_EXP 30.0f

#define LDSB_PITCH 40            // phase-1 B tile row pitch (halves)
#define GW 8                     // persistent-kernel workgroups
#define WPITCH 1032              // resident Wo tile row pitch (halves); 128*1032*2 = 264192 B
#define LDSW_BYTES (128 * WPITCH * 2)

typedef __attribute__((ext_vector_type(16))) _Float16 v16h;
typedef __attribute__((ext_vector_type(8)))  _Float16 v8h;
typedef __attribute__((ext_vector_type(4)))  _Float16 v4h;
typedef __attribute__((ext_vector_type(8)))  float    v8f;
typedef int v4i_async __attribute__((vector_size(16)));

static __device__ __forceinline__ v16h cat8(v8h lo, v8h hi) {
    return __builtin_shufflevector(lo, hi, 0,1,2,3,4,5,6,7,8,9,10,11,12,13,14,15);
}

static __device__ __forceinline__ float softplus_f(float x) {
    return (x > 20.0f) ? x : log1pf(__expf(x));
}

// ---------------------------------------------------------------------------
// gfx1250 async global->LDS staging (ASYNCcnt path) with safe fallback
// ---------------------------------------------------------------------------
#if defined(__has_builtin)
#  if __has_builtin(__builtin_amdgcn_global_load_async_to_lds_b128)
#    define HAS_ASYNC_B128 1
#  endif
#  if __has_builtin(__builtin_amdgcn_s_wait_asynccnt)
#    define HAS_WAIT_ASYNC 1
#  endif
#endif

static __device__ __forceinline__ void copy_b128_to_lds(const _Float16* gp, _Float16* lp) {
#if defined(HAS_ASYNC_B128)
    __builtin_amdgcn_global_load_async_to_lds_b128(
        (v4i_async*)gp, (v4i_async*)lp, 0, 0);
#else
    *(v8h*)lp = *(const v8h*)gp;
#endif
}

static __device__ __forceinline__ void wait_lds_fill() {
#if defined(HAS_ASYNC_B128)
#  if defined(HAS_WAIT_ASYNC)
    __builtin_amdgcn_s_wait_asynccnt(0);
#  else
    asm volatile("s_wait_asynccnt 0x0" ::: "memory");
#  endif
#endif
}

// cumulative-count grid barrier (device scope, L2 atomics)
static __device__ __forceinline__ void grid_barrier(unsigned* cnt, unsigned* phase) {
    __syncthreads();
    if (threadIdx.x == 0) {
        ++(*phase);
        __hip_atomic_fetch_add(cnt, 1u, __ATOMIC_RELEASE, __HIP_MEMORY_SCOPE_AGENT);
        const unsigned target = (*phase) * (unsigned)GW;
        while (__hip_atomic_load(cnt, __ATOMIC_ACQUIRE, __HIP_MEMORY_SCOPE_AGENT) < target) {
            __builtin_amdgcn_s_sleep(1);
        }
    }
    __syncthreads();
}

// ---------------------------------------------------------------------------
// 0) zero recurrent state + grid-barrier counter (deterministic per launch)
// ---------------------------------------------------------------------------
__global__ void zero_state_kernel(float* __restrict__ st, int n) {
    int i = blockIdx.x * blockDim.x + threadIdx.x;
    if (i < n) st[i] = 0.0f;
}

// ---------------------------------------------------------------------------
// 1) convert four fp32 weight matrices to f16 (Wr,Wk,Wv,Wo contiguous)
// ---------------------------------------------------------------------------
__global__ void cvt_weights_kernel(const float* __restrict__ Wr,
                                   const float* __restrict__ Wk,
                                   const float* __restrict__ Wv,
                                   const float* __restrict__ Wo,
                                   _Float16* __restrict__ Wh) {
    int i = (blockIdx.x * blockDim.x + threadIdx.x) * 4;
    const int n = D_DIM * D_DIM;
    const float* srcs[4] = {Wr, Wk, Wv, Wo};
    int mat = i / n;
    int off = i - mat * n;
    const float4 w = *(const float4*)(srcs[mat] + off);
    _Float16* dst = Wh + (size_t)mat * n + off;
    dst[0] = (_Float16)w.x; dst[1] = (_Float16)w.y;
    dst[2] = (_Float16)w.z; dst[3] = (_Float16)w.w;
}

// ---------------------------------------------------------------------------
// 2) per-d Kalman parameters, computed once
// ---------------------------------------------------------------------------
__global__ void param_kernel(const float* __restrict__ raw_decay,
                             const float* __restrict__ pn_param,
                             const float* __restrict__ on_param,
                             float* __restrict__ lamA,
                             float* __restrict__ qnA,
                             float* __restrict__ rnA) {
    int d = blockIdx.x * blockDim.x + threadIdx.x;
    if (d >= D_DIM) return;
    float rate = fminf(-softplus_f(raw_decay[d]), -EPS_DIV);
    rate = fmaxf(rate, -MAX_EXP);
    lamA[d] = __expf(rate);
    qnA[d] = fminf(fmaxf(softplus_f(pn_param[d]), EPS_DIV), 1000.0f);
    rnA[d] = fminf(fmaxf(softplus_f(on_param[d]), EPS_DIV), 1000.0f);
}

// ---------------------------------------------------------------------------
// 3) rmsnorm over all T*B rows -> xn (f32) and xn_h (f16, for WMMA A-frags)
// ---------------------------------------------------------------------------
__global__ void rmsnorm_kernel(const float* __restrict__ x,
                               const float* __restrict__ g,
                               float* __restrict__ xn,
                               _Float16* __restrict__ xnh) {
    __shared__ float red[8];
    const int row = blockIdx.x;
    const int tid = threadIdx.x;
    const float* xr = x + (size_t)row * D_DIM;
    float4 v = *(const float4*)(xr + tid * 4);
    float ss = v.x * v.x + v.y * v.y + v.z * v.z + v.w * v.w;
    #pragma unroll
    for (int off = 16; off > 0; off >>= 1) ss += __shfl_xor(ss, off, 32);
    if ((tid & 31) == 0) red[tid >> 5] = ss;
    __syncthreads();
    if (tid == 0) {
        float s = 0.f;
        #pragma unroll
        for (int i = 0; i < 8; ++i) s += red[i];
        red[0] = s;
    }
    __syncthreads();
    const float ms = red[0] * (1.0f / D_DIM);
    const float scale = rsqrtf(ms + EPS_DIV);
    const float4 gv = *(const float4*)(g + tid * 4);
    float4 o;
    o.x = v.x * scale * gv.x; o.y = v.y * scale * gv.y;
    o.z = v.z * scale * gv.z; o.w = v.w * scale * gv.w;
    *(float4*)(xn + (size_t)row * D_DIM + tid * 4) = o;
    v4h oh; oh[0]=(_Float16)o.x; oh[1]=(_Float16)o.y; oh[2]=(_Float16)o.z; oh[3]=(_Float16)o.w;
    *(v4h*)(xnh + (size_t)row * D_DIM + tid * 4) = oh;
}

// ---------------------------------------------------------------------------
// 4) batched GEMM: rkv[mat] = xn @ W[mat]^T  (f16 in/out)
//    8 waves; block tile 128x128; wave tile 16x128; async LDS double-buffer.
// ---------------------------------------------------------------------------
__global__ __launch_bounds__(256) void gemm_rkv_kernel(
        const _Float16* __restrict__ xnh,
        const _Float16* __restrict__ Wh,
        _Float16* __restrict__ rkv) {
    __shared__ _Float16 ldsB[2][128 * LDSB_PITCH];

    const int tid  = threadIdx.x;
    const int lane = tid & 31;
    const int wave = tid >> 5;
    const int n    = lane & 15;
    const int hi   = lane >> 4;

    const int tilesN = D_DIM / 128;            // 8
    const int tilesM = TBROWS / 128;           // 128
    const int mat = blockIdx.x / (tilesM * tilesN);
    const int rem = blockIdx.x - mat * (tilesM * tilesN);
    const int bm  = rem / tilesN;
    const int bn  = rem - bm * tilesN;

    const _Float16* A   = xnh + (size_t)(bm * 128 + wave * 16) * D_DIM;
    const _Float16* Bw  = Wh + (size_t)mat * D_DIM * D_DIM + (size_t)(bn * 128) * D_DIM;
    _Float16*       Ot  = rkv + (size_t)mat * TBROWS * D_DIM
                              + (size_t)(bm * 128 + wave * 16) * D_DIM + bn * 128;

    auto fill = [&](int buf, int k0) {
        #pragma unroll
        for (int c = 0; c < 2; ++c) {
            const int chunk = tid + c * 256;
            const int row   = chunk >> 2;
            const int seg   = chunk & 3;
            copy_b128_to_lds(Bw + (size_t)row * D_DIM + k0 + seg * 8,
                             &ldsB[buf][row * LDSB_PITCH + seg * 8]);
        }
    };

    fill(0, 0);

    v8f acc[8] = {};
    for (int k0 = 0; k0 < D_DIM; k0 += 32) {
        const int buf = (k0 >> 5) & 1;
        wait_lds_fill();
        __syncthreads();
        if (k0 + 32 < D_DIM) fill(buf ^ 1, k0 + 32);

        const _Float16* ap = A + (size_t)n * D_DIM + k0;
        v8h al = *(const v8h*)(ap + hi * 8);
        v8h ah = *(const v8h*)(ap + 16 + hi * 8);
        v16h af = cat8(al, ah);
        if (k0 + 32 < D_DIM) __builtin_prefetch(ap + 32, 0, 3);

        #pragma unroll
        for (int j = 0; j < 8; ++j) {
            const _Float16* bp = &ldsB[buf][(j * 16 + n) * LDSB_PITCH];
            v8h b0 = *(const v8h*)(bp + hi * 8);
            v8h b1 = *(const v8h*)(bp + 16 + hi * 8);
            v16h bf = cat8(b0, b1);
            acc[j] = __builtin_amdgcn_wmma_f32_16x16x32_f16(
                false, af, false, bf, (short)0, acc[j], false, false);
        }
    }

    #pragma unroll
    for (int j = 0; j < 8; ++j) {
        #pragma unroll
        for (int i = 0; i < 8; ++i) {
            Ot[(size_t)(i + hi * 8) * D_DIM + j * 16 + n] = (_Float16)acc[j][i];
        }
    }
}

// ---------------------------------------------------------------------------
// 5) persistent recurrence kernel: GW workgroups loop over all T steps.
//    Each workgroup parks its 128-column Wo slice (128x1024 f16 = 256 KB)
//    in LDS once (async DMA) and keeps it resident for all steps.
//    Per step:  A) Kalman update -> y(f16)   B) hpre = xn_t + y@Wo^T (WMMA)
//               C) k-WTA per row (1 wave/row) -> out[t]
//    Grid barrier = cumulative-count L2 atomic + s_sleep spin.
// ---------------------------------------------------------------------------
__global__ __launch_bounds__(256) void recurrence_kernel(
        const _Float16* __restrict__ rkv,
        const float* __restrict__ pos,
        const float* __restrict__ Wtime,
        const float* __restrict__ lamA,
        const float* __restrict__ qnA,
        const float* __restrict__ rnA,
        float* __restrict__ s_num,
        float* __restrict__ s_den,
        float* __restrict__ s_var,
        _Float16* __restrict__ ybuf,
        float* __restrict__ hpre,
        const float* __restrict__ xn,
        const _Float16* __restrict__ Wo16,
        float* __restrict__ out,
        unsigned* __restrict__ barCnt) {
    extern __shared__ _Float16 ldsW[];          // [128][WPITCH]

    const int tid  = threadIdx.x;
    const int wg   = blockIdx.x;                // 0..GW-1
    const int lane = tid & 31;
    const int wave = tid >> 5;
    const int n    = lane & 15;
    const int hi   = lane >> 4;
    unsigned phase = 0;

    // ---- park this workgroup's Wo slice (rows wg*128..+128, all K) in LDS ----
    {
        const _Float16* Wslice = Wo16 + (size_t)(wg * 128) * D_DIM;
        for (int c = 0; c < 64; ++c) {
            const int chunk = tid + c * 256;            // 0..16383 (16B chunks)
            const int row   = chunk >> 7;               // 128 chunks per row
            const int seg   = chunk & 127;
            copy_b128_to_lds(Wslice + (size_t)row * D_DIM + seg * 8,
                             &ldsW[row * WPITCH + seg * 8]);
        }
        wait_lds_fill();
        __syncthreads();
    }

    const _Float16* rbBase = rkv;
    const _Float16* kbBase = rkv + (size_t)1 * TBROWS * D_DIM;
    const _Float16* vbBase = rkv + (size_t)2 * TBROWS * D_DIM;

    for (int t = 0; t < T_DIM; ++t) {
        const size_t base = (size_t)t * BD;

        // ================= stage A: Kalman update -> y =================
        {
            const _Float16* rb = rbBase + base;
            const _Float16* kb = kbBase + base;
            const _Float16* vb = vbBase + base;
            #pragma unroll 4
            for (int i = 0; i < 32; ++i) {
                const int idx = wg * (BD / GW) + i * 256 + tid;
                const int d = idx & (D_DIM - 1);
                const float lam = lamA[d];
                const float q_noise = qnA[d];
                const float r_noise = rnA[d];

                const float rpre = (float)rb[idx];
                const float kf   = (float)kb[idx];
                float vf         = (float)vb[idx];
                if (d < 2 * MTIME) {
                    const int mm = d >> 1;
                    const int rowbase = idx - d;
                    const float a  = (float)vb[rowbase + 2 * mm];
                    const float bb = (float)vb[rowbase + 2 * mm + 1];
                    const float theta = pos[t] * Wtime[mm];
                    float c, s;
                    __sincosf(theta, &s, &c);
                    vf = (d & 1) ? (a * s + bb * c) : (a * c - bb * s);
                }
                const float w        = __expf(fminf(kf, MAX_EXP));
                const float obs_var  = __expf(fminf(-kf, MAX_EXP)) * r_noise;
                const float prior_num = s_num[idx] * lam;
                const float prior_den = s_den[idx] * lam;
                const float prior_var = s_var[idx] * lam * lam + q_noise;
                const float pred  = prior_num / (prior_den + EPS_DIV);
                const float resid = w * vf - pred;
                const float gain  = prior_var / (prior_var + obs_var + EPS_DIV);
                const float ns    = pred + gain * resid;
                s_num[idx] = ns * (prior_den + w);
                s_den[idx] = prior_den + w;
                s_var[idx] = (1.0f - gain) * prior_var;
                const float r = 1.0f / (1.0f + __expf(-rpre));
                ybuf[idx] = (_Float16)(r * ns);
            }
        }
        grid_barrier(barCnt, &phase);

        // ===== stage B: hpre = xn_t + y @ Wo^T  (B = resident LDS tile) =====
        {
            const int mrow = (wave & 3) * 16;      // M strip (64 rows total)
            const int wc   = wave >> 2;            // N half within 128 cols
            const _Float16* A = ybuf + (size_t)mrow * D_DIM;
            const float* xn_t = xn + base;

            v8f acc[4] = {};
            for (int k0 = 0; k0 < D_DIM; k0 += 32) {
                const _Float16* ap = A + (size_t)n * D_DIM + k0;
                v8h al = *(const v8h*)(ap + hi * 8);
                v8h ah = *(const v8h*)(ap + 16 + hi * 8);
                v16h af = cat8(al, ah);

                #pragma unroll
                for (int j = 0; j < 4; ++j) {
                    const _Float16* bp = &ldsW[(wc * 64 + j * 16 + n) * WPITCH + k0];
                    v8h b0 = *(const v8h*)(bp + hi * 8);
                    v8h b1 = *(const v8h*)(bp + 16 + hi * 8);
                    v16h bf = cat8(b0, b1);
                    acc[j] = __builtin_amdgcn_wmma_f32_16x16x32_f16(
                        false, af, false, bf, (short)0, acc[j], false, false);
                }
            }
            #pragma unroll
            for (int j = 0; j < 4; ++j) {
                #pragma unroll
                for (int i = 0; i < 8; ++i) {
                    const size_t o = (size_t)(mrow + i + hi * 8) * D_DIM
                                   + wg * 128 + wc * 64 + j * 16 + n;
                    hpre[o] = xn_t[o] + acc[j][i];
                }
            }
        }
        grid_barrier(barCnt, &phase);

        // ============ stage C: k-WTA, one wave per row ============
        {
            const int b = wg * 8 + wave;           // 0..63
            const float* hr = hpre + (size_t)b * D_DIM;
            unsigned oi[32];
            #pragma unroll
            for (int i = 0; i < 32; ++i) {
                const unsigned u = __float_as_uint(hr[lane + i * 32]);
                oi[i] = (u & 0x80000000u) ? ~u : (u | 0x80000000u);
            }
            unsigned thr = 0;
            for (int bit = 31; bit >= 0; --bit) {
                const unsigned cand = thr | (1u << bit);
                int c = 0;
                #pragma unroll
                for (int i = 0; i < 32; ++i) c += (oi[i] >= cand);
                #pragma unroll
                for (int off = 16; off > 0; off >>= 1) c += __shfl_xor(c, off, 32);
                if (c >= KWTA_K) thr = cand;
            }
            float* ot = out + base + (size_t)b * D_DIM;
            #pragma unroll
            for (int i = 0; i < 32; ++i) {
                const unsigned u = (oi[i] & 0x80000000u) ? (oi[i] ^ 0x80000000u) : ~oi[i];
                ot[lane + i * 32] = (oi[i] >= thr) ? __uint_as_float(u) : 0.0f;
            }
        }
        grid_barrier(barCnt, &phase);
    }
}

// ---------------------------------------------------------------------------
// host-side orchestration (graph-capture safe: stream launches only)
// ---------------------------------------------------------------------------
extern "C" void kernel_launch(void* const* d_in, const int* in_sizes, int n_in,
                              void* d_out, int out_size, void* d_ws, size_t ws_size,
                              hipStream_t stream) {
    const float* x_seq     = (const float*)d_in[0];
    const float* pos       = (const float*)d_in[1];
    const float* Wr        = (const float*)d_in[2];
    const float* Wk        = (const float*)d_in[3];
    const float* Wv        = (const float*)d_in[4];
    const float* Wo        = (const float*)d_in[5];
    const float* norm_w    = (const float*)d_in[6];
    const float* raw_decay = (const float*)d_in[7];
    const float* pn_param  = (const float*)d_in[8];
    const float* on_param  = (const float*)d_in[9];
    const float* W_time    = (const float*)d_in[10];
    float* out = (float*)d_out;

    constexpr size_t SZ_W   = 4ull * D_DIM * D_DIM * sizeof(_Float16);   // 8 MB
    constexpr size_t SZ_XN  = (size_t)TBROWS * D_DIM * sizeof(float);    // 64 MB
    constexpr size_t SZ_XNH = (size_t)TBROWS * D_DIM * sizeof(_Float16); // 32 MB
    constexpr size_t SZ_RKV = 3ull * TBROWS * D_DIM * sizeof(_Float16);  // 96 MB
    constexpr size_t OFF_W   = 0;
    constexpr size_t OFF_XN  = OFF_W + SZ_W;
    constexpr size_t OFF_XNH = OFF_XN + SZ_XN;
    constexpr size_t OFF_RKV = OFF_XNH + SZ_XNH;
    constexpr size_t OFF_ST  = OFF_RKV + SZ_RKV;                         // 3*BD f32 + bar
    constexpr size_t OFF_Y   = OFF_ST + (3ull * BD + 64) * sizeof(float);
    constexpr size_t OFF_H   = OFF_Y + (size_t)BD * sizeof(_Float16);
    constexpr size_t OFF_PAR = OFF_H + (size_t)BD * sizeof(float);       // lam,qn,rn

    char* ws = (char*)d_ws;
    _Float16* Wh   = (_Float16*)(ws + OFF_W);
    float*    xn   = (float*)   (ws + OFF_XN);
    _Float16* xnh  = (_Float16*)(ws + OFF_XNH);
    _Float16* rkv  = (_Float16*)(ws + OFF_RKV);
    float*    sNum = (float*)   (ws + OFF_ST);
    float*    sDen = sNum + BD;
    float*    sVar = sDen + BD;
    unsigned* barCnt = (unsigned*)(sNum + 3 * BD);
    _Float16* ybuf = (_Float16*)(ws + OFF_Y);
    float*    hpre = (float*)   (ws + OFF_H);
    float*    lamA = (float*)   (ws + OFF_PAR);
    float*    qnA  = lamA + D_DIM;
    float*    rnA  = qnA + D_DIM;

    // state + barrier counter zeroed every call (deterministic)
    zero_state_kernel<<<(3 * BD + 64 + 255) / 256, 256, 0, stream>>>(sNum, 3 * BD + 64);
    cvt_weights_kernel<<<(4 * D_DIM * D_DIM / 4 + 255) / 256, 256, 0, stream>>>(
        Wr, Wk, Wv, Wo, Wh);
    param_kernel<<<(D_DIM + 255) / 256, 256, 0, stream>>>(
        raw_decay, pn_param, on_param, lamA, qnA, rnA);

    rmsnorm_kernel<<<TBROWS, 256, 0, stream>>>(x_seq, norm_w, xn, xnh);
    {
        const int blocks = 3 * (TBROWS / 128) * (D_DIM / 128);   // 3072
        gemm_rkv_kernel<<<blocks, 256, 0, stream>>>(xnh, Wh, rkv);
    }

    const _Float16* Wo16 = Wh + (size_t)3 * D_DIM * D_DIM;
    static bool attr_set = false;
    if (!attr_set) {
        hipFuncSetAttribute((const void*)recurrence_kernel,
                            hipFuncAttributeMaxDynamicSharedMemorySize, LDSW_BYTES);
        attr_set = true;
    }
    recurrence_kernel<<<GW, 256, LDSW_BYTES, stream>>>(
        rkv, pos, W_time, lamA, qnA, rnA,
        sNum, sDen, sVar, ybuf, hpre, xn, Wo16, out, barCnt);
}
